// Attention_16947940950479
// MI455X (gfx1250) — matrix operand order repaired and checked by reading, not compiled, not run on hardware
//
#include <hip/hip_runtime.h>
#include <cstdint>
#include <cstddef>

// ---------------------------------------------------------------------------
// Types
// ---------------------------------------------------------------------------
typedef unsigned short u16;
typedef u16   u16x4   __attribute__((ext_vector_type(4)));
typedef u16   u16x8   __attribute__((ext_vector_type(8)));
typedef __bf16 bf16x16 __attribute__((ext_vector_type(16)));
typedef float f32x8   __attribute__((ext_vector_type(8)));
typedef int   i32x4   __attribute__((__vector_size__(16)));

// Problem constants (match the reference)
constexpr int BB = 4;      // batch
constexpr int SS = 2048;   // sequence
constexpr int FF = 1024;   // features
constexpr int DK = 1024;   // head dim

#if defined(__HIP_DEVICE_COMPILE__) && __has_builtin(__builtin_amdgcn_global_load_async_to_lds_b128)
#define HAVE_ASYNC_LDS 1
#else
#define HAVE_ASYNC_LDS 0
#endif

__device__ __forceinline__ u16 f32_to_bf16_rne(float f) {
  unsigned u = __float_as_uint(f);
  u += 0x7FFFu + ((u >> 16) & 1u);   // round-to-nearest-even
  return (u16)(u >> 16);
}

#if HAVE_ASYNC_LDS
// GLOBAL_LOAD_ASYNC_TO_LDS_B128: IOFFSET applies to BOTH global and LDS addr.
// Builtin signature (from hipcc diagnostic): (v4i AS(1)*, v4i AS(3)*, Ii, Ii).
typedef __attribute__((address_space(1))) i32x4 gi32x4;
typedef __attribute__((address_space(3))) i32x4 li32x4;

template<int OFF>
__device__ __forceinline__ void async_ld128(const u16* g, u16* l) {
  __builtin_amdgcn_global_load_async_to_lds_b128(
      (gi32x4*)(void*)g, (li32x4*)(void*)l, OFF, 0);
}
__device__ __forceinline__ void wait_async0() {
  asm volatile("s_wait_asynccnt 0" ::: "memory");
}
#endif

// ---------------------------------------------------------------------------
// f32 -> bf16 conversion (vectorized 4/thread; all counts divide exactly)
// ---------------------------------------------------------------------------
__global__ __launch_bounds__(256) void cvt_f32_to_bf16(
    const float* __restrict__ in, u16* __restrict__ out) {
  const size_t i = (((size_t)blockIdx.x * blockDim.x) + threadIdx.x) << 2;
  const float4 v = *(const float4*)(in + i);
  u16x4 o;
  o[0] = f32_to_bf16_rne(v.x);
  o[1] = f32_to_bf16_rne(v.y);
  o[2] = f32_to_bf16_rne(v.z);
  o[3] = f32_to_bf16_rne(v.w);
  *(u16x4*)(out + i) = o;
}

// ---------------------------------------------------------------------------
// WMMA GEMM:  C[m,n] = scale * sum_k A[m,k] * Bm[n,k]  (+ bias[n])
//   A  : bf16 row-major [M,K], leading dim lda, batch stride strideA
//   Bm : bf16 row-major [N,K], leading dim ldb, batch stride strideB
//   C  : f32 or bf16; TRANS_OUT writes C[n*ldc + m] (bf16 only, for V^T)
// Block: 256 threads = 8 wave32; tile 128x128, K-step 32, double-buffered LDS
// filled via async global->LDS (ASYNCcnt) when available.
// ---------------------------------------------------------------------------
template<bool OUT_BF16, bool TRANS_OUT>
__global__ __launch_bounds__(256) void gemm_wmma_bf16(
    const u16* __restrict__ A,  int lda, long long strideA,
    const u16* __restrict__ Bm, int ldb, long long strideB,
    void* __restrict__ Cvoid,   int ldc, long long strideC,
    const float* __restrict__ bias, float scale, int K) {

  __shared__ alignas(16) u16 shA[2][128][40];   // 32 cols + 8 pad (bank spread)
  __shared__ alignas(16) u16 shB[2][128][40];

  const int tid  = threadIdx.x;
  const int lane = tid & 31;
  const int wave = tid >> 5;
  const int wm   = wave >> 1;          // 0..3 : 32-row group
  const int wn   = wave & 1;           // 0..1 : 64-col group

  const long long bz = blockIdx.z;
  A  += bz * strideA;
  Bm += bz * strideB;

  const int row0 = blockIdx.y * 128;
  const int col0 = blockIdx.x * 128;

  // Global -> LDS loader: each thread moves 16 bf16 of A and of B per K-step.
  const int lrow = tid >> 1;           // 0..127
  const int lcol = (tid & 1) << 4;     // 0 or 16
  const u16* gA = A  + (long long)(row0 + lrow) * lda + lcol;
  const u16* gB = Bm + (long long)(col0 + lrow) * ldb + lcol;

  f32x8 acc[2][4];
#pragma unroll
  for (int mi = 0; mi < 2; ++mi)
#pragma unroll
    for (int ni = 0; ni < 4; ++ni)
      acc[mi][ni] = (f32x8){0.f,0.f,0.f,0.f,0.f,0.f,0.f,0.f};

  const int KT = K >> 5;

  auto load_tile = [&](int buf, int k0) {
#if HAVE_ASYNC_LDS
    // VGPR-free staging path: ASYNCcnt-tracked direct-to-LDS copies.
    async_ld128<0> (gA + k0, &shA[buf][lrow][lcol]);
    async_ld128<16>(gA + k0, &shA[buf][lrow][lcol]);
    async_ld128<0> (gB + k0, &shB[buf][lrow][lcol]);
    async_ld128<16>(gB + k0, &shB[buf][lrow][lcol]);
#else
    const uint4* pa = (const uint4*)(gA + k0);
    const uint4* pb = (const uint4*)(gB + k0);
    uint4 a0 = pa[0], a1 = pa[1];
    uint4 b0 = pb[0], b1 = pb[1];
    *(uint4*)&shA[buf][lrow][lcol]     = a0;
    *(uint4*)&shA[buf][lrow][lcol + 8] = a1;
    *(uint4*)&shB[buf][lrow][lcol]     = b0;
    *(uint4*)&shB[buf][lrow][lcol + 8] = b1;
#endif
  };

  load_tile(0, 0);
#if HAVE_ASYNC_LDS
  wait_async0();
#endif
  __syncthreads();

  // ISA fragment layouts (CDNA5 ISA §7.12.2), wave32:
  //  A 16x32 bf16: lane<16 -> row=lane,    K = {0..7, 16..23}
  //                lane>=16 -> row=lane-16, K = {8..15, 24..31}
  //  B 32x16 bf16: lane<16 -> col=lane,    K = {0..7, 16..23} ; lane>=16 -> K = {8..15, 24..31}
  const int ar = lane & 15;
  const int ak = (lane >> 4) << 3;     // 0 or 8
  const int bk = (lane >> 4) << 3;     // 0 or 8

  for (int kt = 0; kt < KT; ++kt) {
    const int buf = kt & 1;
    if (kt + 2 < KT) {                         // -> global_prefetch_b8
      __builtin_prefetch(gA + ((kt + 2) << 5), 0, 1);
      __builtin_prefetch(gB + ((kt + 2) << 5), 0, 1);
    }
    if (kt + 1 < KT) load_tile(buf ^ 1, (kt + 1) << 5);

    bf16x16 af[2];
#pragma unroll
    for (int mi = 0; mi < 2; ++mi) {
      const u16* p = &shA[buf][wm * 32 + mi * 16 + ar][0];
      u16x8 lo = *(const u16x8*)(p + ak);
      u16x8 hi = *(const u16x8*)(p + ak + 16);
      af[mi] = __builtin_bit_cast(bf16x16,
          __builtin_shufflevector(lo, hi, 0,1,2,3,4,5,6,7,8,9,10,11,12,13,14,15));
    }
    bf16x16 bfm[4];
#pragma unroll
    for (int ni = 0; ni < 4; ++ni) {
      const u16* p = &shB[buf][wn * 64 + ni * 16 + ar][0];
      u16x8 lo = *(const u16x8*)(p + bk);
      u16x8 hi = *(const u16x8*)(p + bk + 16);
      bfm[ni] = __builtin_bit_cast(bf16x16,
          __builtin_shufflevector(lo, hi, 0,1,2,3,4,5,6,7,8,9,10,11,12,13,14,15));
    }
#pragma unroll
    for (int mi = 0; mi < 2; ++mi)
#pragma unroll
      for (int ni = 0; ni < 4; ++ni)
        acc[mi][ni] = __builtin_amdgcn_wmma_f32_16x16x32_bf16(
            false, af[mi], false, bfm[ni], (short)0, acc[mi][ni], false, false);

#if HAVE_ASYNC_LDS
    wait_async0();        // this wave's LDS fills for buf^1 are complete
#endif
    __syncthreads();      // all waves' fills visible; reads of buf done
  }

  // Epilogue. C frag layout: VGPR r, lane<16 -> M=r,N=lane ; lane>=16 -> M=8+r.
  const int mhi = (lane >> 4) << 3;
  const int nlo = lane & 15;
  const bool has_bias = (bias != nullptr);
#pragma unroll
  for (int mi = 0; mi < 2; ++mi) {
#pragma unroll
    for (int ni = 0; ni < 4; ++ni) {
#pragma unroll
      for (int r = 0; r < 8; ++r) {
        const int gm = row0 + wm * 32 + mi * 16 + mhi + r;
        const int gn = col0 + wn * 64 + ni * 16 + nlo;
        float v = acc[mi][ni][r] * scale;
        if (has_bias) v += bias[gn];
        if constexpr (OUT_BF16) {
          u16* C = (u16*)Cvoid + bz * strideC;
          if constexpr (TRANS_OUT)
            C[(long long)gn * ldc + gm] = f32_to_bf16_rne(v);
          else
            C[(long long)gm * ldc + gn] = f32_to_bf16_rne(v);
        } else {
          float* C = (float*)Cvoid + bz * strideC;
          C[(long long)gm * ldc + gn] = v;
        }
      }
    }
  }
}

// ---------------------------------------------------------------------------
// Softmax over the QUERY axis (axis=-2): for each key column k, normalize
// over q. Lane owns one column -> every global access is coalesced.
// ---------------------------------------------------------------------------
__global__ __launch_bounds__(256) void col_softmax_bf16(
    const float* __restrict__ scores, u16* __restrict__ attn) {
  const int k = blockIdx.x * blockDim.x + threadIdx.x;  // 0..SS-1
  const long long base = (long long)blockIdx.y * SS * SS + k;
  const float* sp = scores + base;
  float m = -3.4e38f, s = 0.f;
  for (int q = 0; q < SS; ++q) {
    const float v = sp[(long long)q * SS];
    const float nm = fmaxf(m, v);
    s = s * __expf(m - nm) + __expf(v - nm);
    m = nm;
  }
  const float inv = __frcp_rn(s);
  u16* ap = attn + base;
  for (int q = 0; q < SS; ++q) {
    const float v = sp[(long long)q * SS];
    ap[(long long)q * SS] = f32_to_bf16_rne(__expf(v - m) * inv);
  }
}

// ---------------------------------------------------------------------------
// Host-side orchestration
// ---------------------------------------------------------------------------
extern "C" void kernel_launch(void* const* d_in, const int* in_sizes, int n_in,
                              void* d_out, int out_size, void* d_ws, size_t ws_size,
                              hipStream_t stream) {
  (void)in_sizes; (void)n_in; (void)out_size; (void)ws_size;

  const float* x  = (const float*)d_in[0];
  const float* Wq = (const float*)d_in[1];
  const float* bq = (const float*)d_in[2];
  const float* Wk = (const float*)d_in[3];
  const float* bk = (const float*)d_in[4];
  const float* Wv = (const float*)d_in[5];
  const float* bv = (const float*)d_in[6];
  const float* Wo = (const float*)d_in[7];
  const float* bo = (const float*)d_in[8];
  float* out = (float*)d_out;

  const size_t NX = (size_t)BB * SS * FF;   // x elements
  const size_t NW = (size_t)DK * FF;        // weight elements
  const size_t NQ = (size_t)BB * SS * DK;   // q/k/v elements
  const size_t NS = (size_t)BB * SS * SS;   // score elements

  // Workspace layout (~184 MB total)
  u16*   xb   = (u16*)d_ws;
  u16*   wqb  = xb  + NX;
  u16*   wkb  = wqb + NW;
  u16*   wvb  = wkb + NW;
  u16*   wob  = wvb + NW;
  u16*   qb   = wob + NW;
  u16*   kb   = qb  + NQ;
  u16*   vtb  = kb  + NQ;                   // V^T : [DK][BB*SS]
  float* sc   = (float*)(vtb + NQ);         // scores f32 [BB][SS][SS]
  u16*   attn = (u16*)(sc + NS);            // attn bf16
  u16*   ctx  = attn + NS;                  // ctx bf16 [BB*SS][DK]

  const dim3 blk(256);

  // 1) bf16 conversions (4 elems/thread)
  cvt_f32_to_bf16<<<(unsigned)(NX / 1024), blk, 0, stream>>>(x,  xb);
  cvt_f32_to_bf16<<<(unsigned)(NW / 1024), blk, 0, stream>>>(Wq, wqb);
  cvt_f32_to_bf16<<<(unsigned)(NW / 1024), blk, 0, stream>>>(Wk, wkb);
  cvt_f32_to_bf16<<<(unsigned)(NW / 1024), blk, 0, stream>>>(Wv, wvb);
  cvt_f32_to_bf16<<<(unsigned)(NW / 1024), blk, 0, stream>>>(Wo, wob);

  // 2) Q = x·Wq^T + bq  (bf16 out)
  gemm_wmma_bf16<true,false><<<dim3(DK/128, (BB*SS)/128, 1), blk, 0, stream>>>(
      xb, FF, 0, wqb, FF, 0, qb, DK, 0, bq, 1.0f, FF);
  // 3) K = x·Wk^T + bk
  gemm_wmma_bf16<true,false><<<dim3(DK/128, (BB*SS)/128, 1), blk, 0, stream>>>(
      xb, FF, 0, wkb, FF, 0, kb, DK, 0, bk, 1.0f, FF);
  // 4) V = x·Wv^T + bv, stored TRANSPOSED: vtb[d][b*SS+s]
  gemm_wmma_bf16<true,true><<<dim3(DK/128, (BB*SS)/128, 1), blk, 0, stream>>>(
      xb, FF, 0, wvb, FF, 0, vtb, BB*SS, 0, bv, 1.0f, FF);

  // 5) scores_b = (Q_b · K_b^T) / sqrt(DK)   (f32 out, batched via grid.z)
  gemm_wmma_bf16<false,false><<<dim3(SS/128, SS/128, BB), blk, 0, stream>>>(
      qb, DK, (long long)SS * DK,
      kb, DK, (long long)SS * DK,
      sc, SS, (long long)SS * SS,
      nullptr, 0.03125f, DK);

  // 6) softmax over query axis -> bf16 attn
  col_softmax_bf16<<<dim3(SS/256, BB), blk, 0, stream>>>(sc, attn);

  // 7) ctx_b = attn_b · V_b  (V supplied as V^T slice: Bm[n,k]=V_b[k,n])
  gemm_wmma_bf16<true,false><<<dim3(DK/128, SS/128, BB), blk, 0, stream>>>(
      attn, SS,    (long long)SS * SS,
      vtb,  BB*SS, (long long)SS,
      ctx,  DK,    (long long)SS * DK,
      nullptr, 1.0f, SS);

  // 8) out = ctx · Wo^T + bo  (f32 out)
  gemm_wmma_bf16<false,false><<<dim3(FF/128, (BB*SS)/128, 1), blk, 0, stream>>>(
      ctx, DK, 0, wob, DK, 0, out, FF, 0, bo, 1.0f, DK);
}
